// LocalAggregation_17214228922721
// MI455X (gfx1250) — compile-verified, matrix operand order checked by reading, not run on hardware
//
#include <hip/hip_runtime.h>
#include <cstdint>

// ---- problem constants (from reference) ----
#define BB   4
#define NN   4096
#define KK   32
#define CC   192          // channels (contraction 192 = 6 * 32) + 3 dp channels peeled
#define CP3  195
#define PTS  8            // points per block
#define EPSV 1e-5f

typedef __attribute__((ext_vector_type(16))) _Float16 v16h;
typedef __attribute__((ext_vector_type(2)))  _Float16 h2;
typedef __attribute__((ext_vector_type(8)))  float    v8f;
typedef __attribute__((ext_vector_type(4)))  unsigned int u32x4;
typedef __attribute__((ext_vector_type(8)))  int      i32x8;
typedef __attribute__((ext_vector_type(4)))  int      i32x4;

// ---- workspace layout (contiguous 77,568 B, TDM-copied to LDS base) ----
#define WS_WFRAG 0          // f16 A-fragment-swizzled W[:,3:195] (192x192 halves)
#define WS_W3    73728      // f32 W[:,0:3]  (192*3)
#define WS_INV   76032      // f32 gamma/sqrt(var+eps) (192)
#define WS_B2    76800      // f32 beta - rmean*inv    (192)
#define WS_BYTES 77568

// ---- LDS layout (dynamic shared) ----
#define L_WFRAG  0
#define L_W3     73728
#define L_INV    76032
#define L_B2     76800
#define L_HB     77568   // 12 B-fragments * 512 halves * 2B = 12288
#define L_DP     89856   // 3*32 f32 = 384
#define L_IDX    90240   // 32 i32  = 128
#define L_DIM    90368   // 32 f32  = 128 (500^(-j/32))
#define L_YRED   90496   // 192*PTS i32 = 6144
#define L_BYTES  96640

// A/B fragment half-index pattern (ISA 7.12.2, 16-bit 16x32):
//  lane L holds row/col = L%16 ; half i -> k_rel = (i<8 ? i : i+8) + 8*(L>=16)

__global__ void __launch_bounds__(256)
la_prep(const float* __restrict__ W, const float* __restrict__ gamma,
        const float* __restrict__ beta, const float* __restrict__ rmean,
        const float* __restrict__ rvar, unsigned char* __restrict__ ws)
{
    int q = (int)blockIdx.x * 256 + (int)threadIdx.x;
    if (q < 36864) {                       // Wfrag halves
        int f = q >> 9, rem = q & 511, L = rem >> 4, i = rem & 15;
        int Mt = f / 6, Kc = f % 6;
        int m = Mt * 16 + (L & 15);
        int k = Kc * 32 + ((i < 8 ? i : i + 8) + ((L >= 16) ? 8 : 0));
        ((_Float16*)(ws + WS_WFRAG))[q] = (_Float16)W[m * CP3 + 3 + k];
    } else if (q < 36864 + 576) {          // W[:,0:3]
        int t = q - 36864;
        ((float*)(ws + WS_W3))[t] = W[(t / 3) * CP3 + (t % 3)];
    } else if (q < 36864 + 576 + 192) {    // BN fold
        int m = q - (36864 + 576);
        float inv = gamma[m] * rsqrtf(rvar[m] + EPSV);
        ((float*)(ws + WS_INV))[m] = inv;
        ((float*)(ws + WS_B2))[m]  = beta[m] - rmean[m] * inv;
    }
}

__global__ void __launch_bounds__(256)
la_main(const float* __restrict__ p, const float* __restrict__ x,
        const int* __restrict__ idx, const unsigned char* __restrict__ ws,
        float* __restrict__ out)
{
    extern __shared__ unsigned char smem[];
    const int tid  = (int)threadIdx.x;
    const int wave = tid >> 5, lane = tid & 31;

    _Float16* Wfrag = (_Float16*)(smem + L_WFRAG);
    float*    w3L   = (float*)(smem + L_W3);
    float*    invL  = (float*)(smem + L_INV);
    float*    b2L   = (float*)(smem + L_B2);
    _Float16* hB    = (_Float16*)(smem + L_HB);
    float*    dpL   = (float*)(smem + L_DP);
    int*      idxL  = (int*)(smem + L_IDX);
    float*    dimL  = (float*)(smem + L_DIM);
    int*      yredI = (int*)(smem + L_YRED);

    // ---- TDM bulk copy: ws[0..77568) -> LDS[0..77568) (wave 0 issues) ----
    if (wave == 0) {
        unsigned ldsBase = (unsigned)(uintptr_t)smem;          // low 32b of flat LDS addr = LDS offset
        unsigned long long ga = (unsigned long long)(uintptr_t)ws;
        u32x4 g0 = {};
        g0[0] = 1u;                                            // count=1 valid descriptor
        g0[1] = ldsBase;                                       // lds_addr [63:32]
        g0[2] = (unsigned)ga;                                  // global_addr lo
        g0[3] = ((unsigned)(ga >> 32) & 0x1FFFFFFu) | (2u << 30); // addr hi | type=2
        unsigned td0 = WS_BYTES / 8;                           // 9696 elems of 8B
        i32x8 g1 = {};
        g1[0] = (int)(3u << 16);                               // data_size=3 (8B)
        g1[1] = (int)(td0 << 16);                              // tensor_dim0[15:0] @ dw1[31:16]
        g1[2] = (int)((td0 >> 16) | (1u << 16));               // dim0 hi | tensor_dim1=1
        g1[3] = (int)(td0 << 16);                              // tile_dim0 @ dw3[31:16]
        g1[5] = (int)td0;                                      // tensor_dim0_stride lo32
        i32x4 z4 = {};
#if __clang_major__ >= 23
        i32x8 z8 = {};
        __builtin_amdgcn_tensor_load_to_lds(g0, g1, z4, z4, z8, 0);
#else
        __builtin_amdgcn_tensor_load_to_lds(g0, g1, z4, z4, 0);
#endif
        __builtin_amdgcn_s_wait_tensorcnt(0);
    }
    if (wave == 1) dimL[lane] = __expf(-0.19420717f * (float)lane); // 500^(-t/32)
    __syncthreads();

    const int blk = (int)blockIdx.x;
    const int b   = blk >> 9;                 // NN/PTS = 512 blocks per batch
    const int n0  = (blk & 511) * PTS;

    // ---- hoist A fragments (this wave's 3 M-tiles x 6 K-chunks) into VGPRs ----
    const int MtBase = 3 * (wave >> 1);       // waves {2q,2q+1} share Mt {3q..3q+2}
    const int Nt     = wave & 1;              // fixed neighbor half per wave
    v16h A[3][6];
#pragma unroll
    for (int u = 0; u < 3; ++u)
#pragma unroll
        for (int kc = 0; kc < 6; ++kc)
            A[u][kc] = *(const v16h*)(Wfrag + (((MtBase + u) * 6 + kc) * 32 + lane) * 16);

    const int nr = lane & 15, hi = lane >> 4;

    for (int pt = 0; pt < PTS; ++pt) {
        const int n = n0 + pt;
        if (tid < KK) {
            int iv = idx[(b * NN + n) * KK + tid];
            idxL[tid] = iv;
            float cx = p[(b * NN + n) * 3 + 0];
            float cy = p[(b * NN + n) * 3 + 1];
            float cz = p[(b * NN + n) * 3 + 2];
            dpL[0 * 32 + tid] = p[(b * NN + iv) * 3 + 0] - cx;
            dpL[1 * 32 + tid] = p[(b * NN + iv) * 3 + 1] - cy;
            dpL[2 * 32 + tid] = p[(b * NN + iv) * 3 + 2] - cz;
        }
        if (tid < CC) yredI[pt * CC + tid] = 0;   // ReLU output >= 0
        __syncthreads();

        // ---- sin/cos embedding -> packed f16 pairs in B-fragment layout ----
        // channel pair (c, c+1): same lane L, adjacent halfwords (only bit0 of
        // k_rel differs) -> one 4-byte DS store. cos(x) = sin(x + pi/2) keeps
        // the loop branch-free (no exec-mask splits).
        for (int q = tid; q < (CC / 2) * KK; q += 256) {   // 3072 / 256 = 12, exact
            int cp = q >> 5;                 // 0..95  (channel pair)
            int kk = q & 31;                 // neighbor
            int c  = cp * 2;                 // even channel
            int d  = c >> 6;                 // coord axis
            int j  = c & 63;                 // even; pair never straddles sin|cos
            int jj = j & 31;
            float dpv   = 50.0f * dpL[d * 32 + kk];
            float phase = (j >= 32) ? 1.57079632679f : 0.0f;   // cos = sin(+pi/2)
            float s0 = __sinf(dpv * dimL[jj]     + phase);
            float s1 = __sinf(dpv * dimL[jj + 1] + phase);
            int iv = idxL[kk];
            float x0 = x[(b * CC + c)     * NN + iv];
            float x1 = x[(b * CC + c + 1) * NN + iv];
            h2 hv;
            hv[0] = (_Float16)((x0 + 1.0f) * s0);
            hv[1] = (_Float16)((x1 + 1.0f) * s1);
            int kr = c & 31;                 // even
            int L  = (kk & 15) + ((kr & 8)  ? 16 : 0);
            int i  = (kr & 7)  + ((kr & 16) ?  8 : 0);   // even -> 4B aligned
            *(h2*)(hB + (((c >> 5) * 2 + (kk >> 4)) * 32 + L) * 16 + i) = hv;
        }
        __syncthreads();

        // ---- WMMA: 3 tiles (Mt=MtBase..+2, fixed Nt), contraction 6x32,
        //      B fragment double-buffered so DS latency hides behind WMMAs ----
        v8f acc[3] = {v8f{}, v8f{}, v8f{}};
        v16h bf = *(const v16h*)(hB + ((0 * 2 + Nt) * 32 + lane) * 16);
#pragma unroll
        for (int kc = 0; kc < 6; ++kc) {
            v16h bn = bf;
            if (kc < 5)
                bn = *(const v16h*)(hB + (((kc + 1) * 2 + Nt) * 32 + lane) * 16);
            acc[0] = __builtin_amdgcn_wmma_f32_16x16x32_f16(
                false, A[0][kc], false, bf, (short)0, acc[0], false, false);
            acc[1] = __builtin_amdgcn_wmma_f32_16x16x32_f16(
                false, A[1][kc], false, bf, (short)0, acc[1], false, false);
            acc[2] = __builtin_amdgcn_wmma_f32_16x16x32_f16(
                false, A[2][kc], false, bf, (short)0, acc[2], false, false);
            bf = bn;
        }

        // ---- epilogue: dp channels + BN + ReLU, cross-lane max, sparse atomics ----
        const int nn  = Nt * 16 + nr;
        const float d0 = dpL[nn], d1 = dpL[32 + nn], d2 = dpL[64 + nn];
#pragma unroll
        for (int u = 0; u < 3; ++u) {
            const int Mt    = MtBase + u;
            const int mbase = Mt * 16 + 8 * hi;       // 8 consecutive channels for this half
            // broadcast float4 loads of the per-channel constants (contiguous rows)
            float4 wq[6], iq[2], bq[2];
#pragma unroll
            for (int t = 0; t < 6; ++t) wq[t] = ((const float4*)(w3L + mbase * 3))[t];
#pragma unroll
            for (int t = 0; t < 2; ++t) iq[t] = ((const float4*)(invL + mbase))[t];
#pragma unroll
            for (int t = 0; t < 2; ++t) bq[t] = ((const float4*)(b2L + mbase))[t];
            const float* wrow = (const float*)wq;     // wrow[r*3 + d]
            const float* invv = (const float*)iq;     // invv[r]
            const float* b2v  = (const float*)bq;     // b2v[r]

            float vmax[8];
#pragma unroll
            for (int r = 0; r < 8; ++r) {
                float y = acc[u][r]
                        + wrow[r * 3 + 0] * d0
                        + wrow[r * 3 + 1] * d1
                        + wrow[r * 3 + 2] * d2;
                vmax[r] = fmaxf(y * invv[r] + b2v[r], 0.0f);
            }
            // max over the 16 neighbor columns held by this half (xor stays in half)
#pragma unroll
            for (int r = 0; r < 8; ++r) {
                float v = vmax[r];
                v = fmaxf(v, __shfl_xor(v, 1, 32));
                v = fmaxf(v, __shfl_xor(v, 2, 32));
                v = fmaxf(v, __shfl_xor(v, 4, 32));
                v = fmaxf(v, __shfl_xor(v, 8, 32));
                vmax[r] = v;
            }
            if (nr == 0) {                            // one lane per half commits
#pragma unroll
                for (int r = 0; r < 8; ++r)
                    atomicMax(&yredI[pt * CC + mbase + r], __float_as_int(vmax[r]));
            }
        }
        __syncthreads();
    }

    // ---- coalesced output: channel m writes 8 consecutive points ----
    if (tid < CC) {
        float4 o0, o1;
        o0.x = __int_as_float(yredI[0 * CC + tid]);
        o0.y = __int_as_float(yredI[1 * CC + tid]);
        o0.z = __int_as_float(yredI[2 * CC + tid]);
        o0.w = __int_as_float(yredI[3 * CC + tid]);
        o1.x = __int_as_float(yredI[4 * CC + tid]);
        o1.y = __int_as_float(yredI[5 * CC + tid]);
        o1.z = __int_as_float(yredI[6 * CC + tid]);
        o1.w = __int_as_float(yredI[7 * CC + tid]);
        float* dst = out + ((size_t)b * CC + tid) * NN + n0;
        *(float4*)(dst)     = o0;
        *(float4*)(dst + 4) = o1;
    }
}

extern "C" void kernel_launch(void* const* d_in, const int* in_sizes, int n_in,
                              void* d_out, int out_size, void* d_ws, size_t ws_size,
                              hipStream_t stream) {
    const float* p     = (const float*)d_in[0];
    const float* x     = (const float*)d_in[1];
    const int*   idx   = (const int*)d_in[2];
    const float* W     = (const float*)d_in[3];
    const float* gamma = (const float*)d_in[4];
    const float* beta  = (const float*)d_in[5];
    const float* rmean = (const float*)d_in[6];
    const float* rvar  = (const float*)d_in[7];

    la_prep<<<147, 256, 0, stream>>>(W, gamma, beta, rmean, rvar, (unsigned char*)d_ws);

    const int nblk = BB * (NN / PTS);   // 2048
    la_main<<<nblk, 256, L_BYTES, stream>>>(p, x, idx, (const unsigned char*)d_ws,
                                            (float*)d_out);
}
